// QuantLinear_20203526161177
// MI455X (gfx1250) — 8-bit matrix operand order repaired and checked by reading, compiled, not run on hardware
//
#include <hip/hip_runtime.h>

typedef __attribute__((ext_vector_type(8))) int v8i;
typedef int v4i_vs __attribute__((vector_size(16)));

#define K_DIM 4096
#define M_DIM 4096
#define A_STRIDE 80
#define B_STRIDE 80

#if __has_builtin(__builtin_amdgcn_global_load_async_to_lds_b128)
#define HAVE_ASYNC_LDS_BUILTIN 1
#endif

__device__ __forceinline__ void async_copy16(const signed char* g,
                                             signed char* s) {
#ifdef HAVE_ASYNC_LDS_BUILTIN
  __builtin_amdgcn_global_load_async_to_lds_b128(
      (__attribute__((address_space(1))) v4i_vs*)g,
      (__attribute__((address_space(3))) v4i_vs*)s, 0, 0);
#else
  unsigned lds_off = (unsigned)(uintptr_t)s;
  asm volatile("global_load_async_to_lds_b128 %0, %1, off" ::"v"(lds_off),
               "v"((unsigned long long)(uintptr_t)g)
               : "memory");
#endif
}

__device__ __forceinline__ void wait_async0() {
#if __has_builtin(__builtin_amdgcn_s_wait_asynccnt)
  __builtin_amdgcn_s_wait_asynccnt(0);
#else
  asm volatile("s_wait_asynccnt 0x0" ::: "memory");
#endif
}

__device__ __forceinline__ int q8(float v, float inv, float zp) {
  float q = rintf(fmaf(v, inv, zp));
  q = fminf(127.0f, fmaxf(-128.0f, q));
  return (int)q;
}

__device__ __forceinline__ int pack4(int a, int b, int c, int d) {
  return (a & 255) | ((b & 255) << 8) | ((c & 255) << 16) | ((d & 255) << 24);
}

__global__ __launch_bounds__(256) void quant_w_kernel(
    const float* __restrict__ W, signed char* __restrict__ Wq,
    float* __restrict__ wscale, int* __restrict__ wsum) {
  __shared__ float rmax[256];
  __shared__ int   rsum[256];
  const int m = blockIdx.x;
  const int t = threadIdx.x;
  const float* row = W + (size_t)m * K_DIM;

  float4 v[4];
#pragma unroll
  for (int i = 0; i < 4; ++i) v[i] = *(const float4*)(row + t * 16 + i * 4);

  float am = 0.0f;
#pragma unroll
  for (int i = 0; i < 4; ++i) {
    am = fmaxf(am, fabsf(v[i].x)); am = fmaxf(am, fabsf(v[i].y));
    am = fmaxf(am, fabsf(v[i].z)); am = fmaxf(am, fabsf(v[i].w));
  }
  rmax[t] = am;
  __syncthreads();
  for (int s = 128; s > 0; s >>= 1) {
    if (t < s) rmax[t] = fmaxf(rmax[t], rmax[t + s]);
    __syncthreads();
  }
  const float absmax = rmax[0];
  const float scale  = (absmax > 0.0f) ? absmax * (1.0f / 127.0f) : 1.0f;
  const float inv    = (absmax > 0.0f) ? 127.0f / absmax : 1.0f;

  int4 packed;
  int sum = 0;
  int q[16];
#pragma unroll
  for (int i = 0; i < 4; ++i) {
    q[i * 4 + 0] = q8(v[i].x, inv, 0.0f);
    q[i * 4 + 1] = q8(v[i].y, inv, 0.0f);
    q[i * 4 + 2] = q8(v[i].z, inv, 0.0f);
    q[i * 4 + 3] = q8(v[i].w, inv, 0.0f);
  }
#pragma unroll
  for (int i = 0; i < 16; ++i) sum += q[i];
  packed.x = pack4(q[0], q[1], q[2], q[3]);
  packed.y = pack4(q[4], q[5], q[6], q[7]);
  packed.z = pack4(q[8], q[9], q[10], q[11]);
  packed.w = pack4(q[12], q[13], q[14], q[15]);
  *(int4*)(Wq + (size_t)m * K_DIM + t * 16) = packed;

  rsum[t] = sum;
  __syncthreads();
  for (int s = 128; s > 0; s >>= 1) {
    if (t < s) rsum[t] += rsum[t + s];
    __syncthreads();
  }
  if (t == 0) {
    wscale[m] = scale;
    wsum[m]   = rsum[0];
  }
}

__global__ __launch_bounds__(256) void quant_x_kernel(
    const float* __restrict__ X, signed char* __restrict__ Xq,
    float* __restrict__ xscale, int* __restrict__ xzp) {
  __shared__ float rmin[256];
  __shared__ float rmax[256];
  const int n = blockIdx.x;
  const int t = threadIdx.x;
  const float* row = X + (size_t)n * K_DIM;

  float4 v[4];
#pragma unroll
  for (int i = 0; i < 4; ++i) v[i] = *(const float4*)(row + t * 16 + i * 4);

  float mn = v[0].x, mx = v[0].x;
#pragma unroll
  for (int i = 0; i < 4; ++i) {
    mn = fminf(mn, fminf(fminf(v[i].x, v[i].y), fminf(v[i].z, v[i].w)));
    mx = fmaxf(mx, fmaxf(fmaxf(v[i].x, v[i].y), fmaxf(v[i].z, v[i].w)));
  }
  rmin[t] = mn;
  rmax[t] = mx;
  __syncthreads();
  for (int s = 128; s > 0; s >>= 1) {
    if (t < s) {
      rmin[t] = fminf(rmin[t], rmin[t + s]);
      rmax[t] = fmaxf(rmax[t], rmax[t + s]);
    }
    __syncthreads();
  }
  const float xmin = rmin[0];
  const float rng  = rmax[0] - xmin;
  const float scale = (rng > 0.0f) ? rng * (1.0f / 255.0f) : 1.0f;
  const float inv   = 1.0f / scale;
  float zpf = rintf(-128.0f - xmin * inv);
  zpf = fminf(127.0f, fmaxf(-128.0f, zpf));

  int4 packed;
  int q[16];
#pragma unroll
  for (int i = 0; i < 4; ++i) {
    q[i * 4 + 0] = q8(v[i].x, inv, zpf);
    q[i * 4 + 1] = q8(v[i].y, inv, zpf);
    q[i * 4 + 2] = q8(v[i].z, inv, zpf);
    q[i * 4 + 3] = q8(v[i].w, inv, zpf);
  }
  packed.x = pack4(q[0], q[1], q[2], q[3]);
  packed.y = pack4(q[4], q[5], q[6], q[7]);
  packed.z = pack4(q[8], q[9], q[10], q[11]);
  packed.w = pack4(q[12], q[13], q[14], q[15]);
  *(int4*)(Xq + (size_t)n * K_DIM + t * 16) = packed;

  if (t == 0) {
    xscale[n] = scale;
    xzp[n]    = (int)zpf;
  }
}

__global__ __launch_bounds__(256) void gemm_i8_kernel(
    const signed char* __restrict__ Xq, const signed char* __restrict__ Wq,
    const float* __restrict__ xs, const int* __restrict__ xzp,
    const float* __restrict__ wsf, const int* __restrict__ wsum,
    const float* __restrict__ bias, float* __restrict__ out) {
  __shared__ signed char smA[2][128 * A_STRIDE];
  __shared__ signed char smB[2][128 * B_STRIDE];

  const int tid    = threadIdx.x;
  const int lane   = tid & 31;
  const int w      = tid >> 5;
  const int wave_n = w >> 2;
  const int wave_m = w & 3;
  const int n0 = blockIdx.y * 128;
  const int m0 = blockIdx.x * 128;

  const int halfsel = lane >> 4;
  const int l16     = lane & 15;

  v8i acc[4][2];
#pragma unroll
  for (int i = 0; i < 4; ++i)
#pragma unroll
    for (int j = 0; j < 2; ++j)
      acc[i][j] = (v8i){0, 0, 0, 0, 0, 0, 0, 0};

  const int fr = tid >> 1;
  const int fh = (tid & 1) * 32;
  const signed char* gA = Xq + (size_t)(n0 + fr) * K_DIM + fh;
  const signed char* gB = Wq + (size_t)(m0 + fr) * K_DIM + fh;

  auto fill = [&](int buf, int k0) {
    signed char* sA = &smA[buf][fr * A_STRIDE + fh];
    signed char* sB = &smB[buf][fr * B_STRIDE + fh];
    async_copy16(gA + k0, sA);
    async_copy16(gA + k0 + 16, sA + 16);
    async_copy16(gB + k0, sB);
    async_copy16(gB + k0 + 16, sB + 16);
  };

  fill(0, 0);
  int cur = 0;
  for (int k0 = 0; k0 < K_DIM; k0 += 64) {
    wait_async0();
    __syncthreads();
    if (k0 + 64 < K_DIM) fill(cur ^ 1, k0 + 64);

    v8i afrag[4];
#pragma unroll
    for (int i = 0; i < 4; ++i) {
      const signed char* p =
          &smA[cur][(wave_n * 64 + 16 * i + l16) * A_STRIDE + halfsel * 8];
      int2 t0 = *(const int2*)(p);
      int2 t1 = *(const int2*)(p + 16);
      int2 t2 = *(const int2*)(p + 32);
      int2 t3 = *(const int2*)(p + 48);
      afrag[i][0] = t0.x; afrag[i][1] = t0.y;
      afrag[i][2] = t1.x; afrag[i][3] = t1.y;
      afrag[i][4] = t2.x; afrag[i][5] = t2.y;
      afrag[i][6] = t3.x; afrag[i][7] = t3.y;
    }
    v8i bfrag[2];
#pragma unroll
    for (int j = 0; j < 2; ++j) {
      const signed char* p =
          &smB[cur][(wave_m * 32 + 16 * j + l16) * B_STRIDE + halfsel * 8];
      int2 t0 = *(const int2*)(p);
      int2 t1 = *(const int2*)(p + 16);
      int2 t2 = *(const int2*)(p + 32);
      int2 t3 = *(const int2*)(p + 48);
      bfrag[j][0] = t0.x; bfrag[j][1] = t0.y;
      bfrag[j][2] = t1.x; bfrag[j][3] = t1.y;
      bfrag[j][4] = t2.x; bfrag[j][5] = t2.y;
      bfrag[j][6] = t3.x; bfrag[j][7] = t3.y;
    }

#pragma unroll
    for (int i = 0; i < 4; ++i)
#pragma unroll
      for (int j = 0; j < 2; ++j)
        acc[i][j] = __builtin_amdgcn_wmma_i32_16x16x64_iu8(
             true, afrag[i],  true, bfrag[j], acc[i][j],
             false,  false);
    cur ^= 1;
  }

#pragma unroll
  for (int i = 0; i < 4; ++i) {
    const int tn_base = n0 + wave_n * 64 + 16 * i + halfsel * 8;
#pragma unroll
    for (int j = 0; j < 2; ++j) {
      const int wm = m0 + wave_m * 32 + 16 * j + l16;
      const float sw = wsf[wm];
      const int   sm = wsum[wm];
      const float bb = bias[wm];
#pragma unroll
      for (int v = 0; v < 8; ++v) {
        const int tn  = tn_base + v;
        const float sx = xs[tn];
        const int   zp = xzp[tn];
        const float val = (float)(acc[i][j][v] - zp * sm) * sx * sw + bb;
        out[(size_t)tn * M_DIM + wm] = val;
      }
    }
  }
}

extern "C" void kernel_launch(void* const* d_in, const int* in_sizes, int n_in,
                              void* d_out, int out_size, void* d_ws,
                              size_t ws_size, hipStream_t stream) {
  const float* x = (const float*)d_in[0];
  const float* W = (const float*)d_in[1];
  const float* b = (const float*)d_in[2];

  const int N = in_sizes[0] / K_DIM;

  char* ws = (char*)d_ws;
  const size_t xq_bytes = (size_t)N * K_DIM;
  const size_t wq_bytes = (size_t)M_DIM * K_DIM;
  signed char* Xq  = (signed char*)ws;
  signed char* Wq  = (signed char*)(ws + xq_bytes);
  float* xs   = (float*)(ws + xq_bytes + wq_bytes);
  int*   xzp  = (int*)(xs + N);
  float* wsf  = (float*)(xzp + N);
  int*   wsum = (int*)(wsf + M_DIM);

  quant_w_kernel<<<M_DIM, 256, 0, stream>>>(W, Wq, wsf, wsum);
  quant_x_kernel<<<N, 256, 0, stream>>>(x, Xq, xs, xzp);

  dim3 grid(M_DIM / 128, N / 128);
  gemm_i8_kernel<<<grid, 256, 0, stream>>>(Xq, Wq, xs, xzp, wsf, wsum, b,
                                           (float*)d_out);
}
